// interactive_LSTM_81896436400336
// MI455X (gfx1250) — matrix operand order repaired and checked by reading, not compiled, not run on hardware
//
#include <hip/hip_runtime.h>

// ---------------------------------------------------------------------------
// interactive LSTM for MI455X (gfx1250), wave32 + v_wmma_f32_16x16x32_bf16
//
// B=64 T=128 M=8 C=H=1024.  Phases:
//   1) weights fp32->bf16 (48MB bf16, L2-resident for all 256 steps)
//   2) separation/grouping -> groups bf16 [16384,1024]
//   3) proj GEMM (WMMA, 64x64 wave tiles): x_proj = A @ Wx^T + bx
//   4) 128 fused step kernels per layer: dual WMMA GEMM (32-row wave tiles,
//      in-register B reuse) + LSTM cell update fused in-register
// ---------------------------------------------------------------------------

#define BB 64
#define TT 128
#define MM 8
#define CC 1024
#define HH 1024
#define G4 4096           // 4*H
#define ROWS_BT (BB*TT*2) // 16384
#define ROWS_ST (BB*2)    // 128

typedef __attribute__((ext_vector_type(4)))  unsigned int u32x4;
typedef __attribute__((ext_vector_type(16))) __bf16       v16bf;
typedef __attribute__((ext_vector_type(8)))  float        v8f;

union FragBF16 { u32x4 q[2]; v16bf v; };

__device__ __forceinline__ unsigned short f2bf(float x) {
  unsigned int u = __float_as_uint(x);
  u += 0x7FFFu + ((u >> 16) & 1u);     // round-to-nearest-even
  return (unsigned short)(u >> 16);
}
__device__ __forceinline__ float sigm(float x) {
  return 1.0f / (1.0f + __expf(-x));
}

// A-matrix 16x32 bf16 fragment. Lanes 0-15 -> M=lane, K=kb..kb+15 contiguous
// (kb = 16*(lane>>4)); two 16B chunks.
__device__ __forceinline__ v16bf load_frag_a(const unsigned short* __restrict__ base,
                                             int row0, int ld, int k0, int lane,
                                             int xor_row) {
  int m   = lane & 15;
  int row = row0 + (m ^ xor_row);           // xor_row=1 swaps group partner rows
  int kb  = k0 + ((lane >> 4) << 4);        // +0 / +16
  const unsigned short* p = base + (size_t)row * ld + kb;
  FragBF16 f;
  f.q[0] = *(const u32x4*)(p);              // K = kb .. kb+7
  f.q[1] = *(const u32x4*)(p + 8);          // K = kb+8 .. kb+15
  return f.v;
}

// B-matrix 32x16 bf16 fragment from ROW-MAJOR W[n][k] (B[k][n] = W[n][k]).
// Lane: N = n0+(lane&15), K = kb..kb+15 contiguous (kb = 16*(lane>>4)).
__device__ __forceinline__ v16bf load_frag_b(const unsigned short* __restrict__ W,
                                             int n0, int ld, int k0, int lane) {
  int n  = n0 + (lane & 15);
  int kb = k0 + ((lane >> 4) << 4);         // +0 / +16
  const unsigned short* p = W + (size_t)n * ld + kb;
  FragBF16 f;
  f.q[0] = *(const u32x4*)(p);
  f.q[1] = *(const u32x4*)(p + 8);
  return f.v;
}

__device__ __forceinline__ v8f wmma_bf16(v16bf a, v16bf b, v8f c) {
  return __builtin_amdgcn_wmma_f32_16x16x32_bf16(false, a, false, b,
                                                 (short)0, c, false, false);
}

// ------------------------------- utilities ---------------------------------
__global__ void f32_to_bf16_kernel(const float* __restrict__ in,
                                   unsigned short* __restrict__ out, int n) {
  for (int i = blockIdx.x * blockDim.x + threadIdx.x; i < n;
       i += gridDim.x * blockDim.x)
    out[i] = f2bf(in[i]);
}
__global__ void add_bias_kernel(const float* __restrict__ a,
                                const float* __restrict__ b,
                                float* __restrict__ out, int n) {
  int i = blockIdx.x * blockDim.x + threadIdx.x;
  if (i < n) out[i] = a[i] + b[i];
}
__global__ void zero_u32_kernel(unsigned int* __restrict__ p, int n) {
  for (int i = blockIdx.x * blockDim.x + threadIdx.x; i < n;
       i += gridDim.x * blockDim.x)
    p[i] = 0u;
}

// --------------------- separation + group pooling (eval) -------------------
__global__ __launch_bounds__(256) void sep_group_kernel(
    const float* __restrict__ feat,    // [B,T,8,1024]
    const float* __restrict__ sep_w,   // [1024]
    const float* __restrict__ sep_b,   // [1]
    unsigned short* __restrict__ groups) { // bf16 [16384,1024]
  int bt = blockIdx.x;
  const float* f0 = feat + (size_t)bt * MM * CC;
  __shared__ float logits[MM];
  int wid = threadIdx.x >> 5, lane = threadIdx.x & 31;

  float part = 0.f;
  const float* fm = f0 + wid * CC;
  for (int c = lane; c < CC; c += 32) part += fm[c] * sep_w[c];
  for (int off = 16; off > 0; off >>= 1) part += __shfl_down(part, off, 32);
  if (lane == 0) logits[wid] = part + sep_b[0];
  __syncthreads();

  float lmax = -1e30f;
  #pragma unroll
  for (int m = 0; m < MM; ++m) lmax = fmaxf(lmax, logits[m]);
  float e[MM], esum = 0.f;
  #pragma unroll
  for (int m = 0; m < MM; ++m) { e[m] = __expf(logits[m] - lmax); esum += e[m]; }
  float sflag[MM], cnt = 0.f;
  #pragma unroll
  for (int m = 0; m < MM; ++m) {
    sflag[m] = (e[m] / esum - 1.0f / MM > 0.f) ? 1.f : 0.f;
    cnt += sflag[m];
  }
  float den0 = cnt + 1e-8f, den1 = ((float)MM - cnt) + 1e-8f;

  size_t orow = (size_t)bt * 2 * CC;
  for (int c = threadIdx.x; c < CC; c += 256) {
    float num0 = 0.f, tot = 0.f;
    #pragma unroll
    for (int m = 0; m < MM; ++m) {
      float v = f0[m * CC + c];
      tot += v;
      num0 += sflag[m] * v;
    }
    groups[orow + c]      = f2bf(num0 / den0);
    groups[orow + CC + c] = f2bf((tot - num0) / den1);
  }
}

// ---------------------- input-projection GEMM (WMMA) -----------------------
// out[16384,4096] = A[16384,1024](bf16) @ W[4096,1024]^T(bf16) + bias.
// Wave tile 64x64: 16 accumulators, 4 A-frags + 4 B-frags -> 16 WMMA per
// k-iter (4x in-register reuse of each fragment, ~4x less L2 traffic).
// Grid = (16384/64)*(4096/64)/4 = 4096 blocks of 128 threads (4 waves).
__global__ __launch_bounds__(128) void gemm_proj_kernel(
    const unsigned short* __restrict__ A,
    const unsigned short* __restrict__ W,
    const float* __restrict__ bias,
    float* __restrict__ out) {
  int wid = threadIdx.x >> 5, lane = threadIdx.x & 31;
  int gw  = blockIdx.x * 4 + wid;          // 0..16383
  int rg  = gw >> 6;                       // row group 0..255 (64 rows)
  int cg  = gw & 63;                       // col group 0..63  (64 cols)
  int row0 = rg * 64;
  int n0   = cg * 64;

  v8f acc[4][4] = {};
  for (int k0 = 0; k0 < HH; k0 += 32) {
    v16bf a[4], b[4];
    #pragma unroll
    for (int i = 0; i < 4; ++i) a[i] = load_frag_a(A, row0 + 16 * i, HH, k0, lane, 0);
    #pragma unroll
    for (int q = 0; q < 4; ++q) b[q] = load_frag_b(W, n0 + 16 * q, HH, k0, lane);
    #pragma unroll
    for (int i = 0; i < 4; ++i)
      #pragma unroll
      for (int q = 0; q < 4; ++q)
        acc[i][q] = wmma_bf16(a[i], b[q], acc[i][q]);
  }

  int nl = lane & 15;
  int mo = (lane >> 4) << 3;               // +0 / +8 row offset inside tile
  #pragma unroll
  for (int q = 0; q < 4; ++q) {
    int n = n0 + 16 * q + nl;
    float bn = bias[n];
    #pragma unroll
    for (int i = 0; i < 4; ++i) {
      int mb = row0 + 16 * i + mo;
      #pragma unroll
      for (int r = 0; r < 8; ++r)
        out[(size_t)(mb + r) * G4 + n] = acc[i][q][r] + bn;
    }
  }
}

// ------------------------ fused recurrent LSTM step ------------------------
// gates = x_proj + h@Wh^T + h_swap@Whh^T + (bh+bhh); cell update fused.
// Wave tile: M=32 (two 16-row tiles) x gate columns {nj, nj+1024, nj+2048,
// nj+3072} -> 8 accumulators; each B fragment feeds 2 row tiles (halves the
// per-step weight re-reads from L2: 16MB * 4 instead of * 8).
// Grid = (128/32)*(1024/16)/4 = 64 blocks of 128 threads (256 waves).
__global__ __launch_bounds__(128) void lstm_step_kernel(
    const float* __restrict__ xproj,          // [16384,4096], row (b*T+t)*2+g
    const unsigned short* __restrict__ Wh,    // bf16 [4096,1024]
    const unsigned short* __restrict__ Whh,   // bf16 [4096,1024]
    const float* __restrict__ bsum,           // [4096] = bh + bhh
    const unsigned short* __restrict__ h_in,  // bf16 [128,1024] (step t-1)
    unsigned short* __restrict__ h_out,       // bf16 [128,1024] (step t)
    float* __restrict__ c_state,              // f32  [128,1024]
    unsigned short* __restrict__ out_bf16,    // layer0: bf16 [16384,1024] or null
    float* __restrict__ out_f32,              // layer1: d_out [B,T,2048] or null
    int t) {
  int wid = threadIdx.x >> 5, lane = threadIdx.x & 31;
  int gw  = blockIdx.x * 4 + wid;           // 0..255
  int rg  = gw >> 6;                        // 0..3  (32-row group)
  int j   = gw & 63;                        // hidden col tile
  int row0 = rg * 32;
  int nj   = j * 16;

  v8f acc[2][4] = {};                       // [row subtile][i,o,f,g]
  for (int k0 = 0; k0 < HH; k0 += 32) {
    v16bf a1[2], a2[2];
    #pragma unroll
    for (int i = 0; i < 2; ++i) {
      a1[i] = load_frag_a(h_in, row0 + 16 * i, HH, k0, lane, 0);  // h[b,g]
      a2[i] = load_frag_a(h_in, row0 + 16 * i, HH, k0, lane, 1);  // h[b,1-g]
    }
    #pragma unroll
    for (int g = 0; g < 4; ++g) {
      v16bf b1 = load_frag_b(Wh, g * HH + nj, HH, k0, lane);
      acc[0][g] = wmma_bf16(a1[0], b1, acc[0][g]);
      acc[1][g] = wmma_bf16(a1[1], b1, acc[1][g]);
      v16bf b2 = load_frag_b(Whh, g * HH + nj, HH, k0, lane);
      acc[0][g] = wmma_bf16(a2[0], b2, acc[0][g]);
      acc[1][g] = wmma_bf16(a2[1], b2, acc[1][g]);
    }
  }

  int nl  = lane & 15;
  int hid = nj + nl;                        // hidden index 0..1023
  int mo  = (lane >> 4) << 3;
  float bs0 = bsum[0 * HH + hid], bs1 = bsum[1 * HH + hid];
  float bs2 = bsum[2 * HH + hid], bs3 = bsum[3 * HH + hid];

  #pragma unroll
  for (int i = 0; i < 2; ++i) {
    int mb = row0 + 16 * i + mo;
    #pragma unroll
    for (int r = 0; r < 8; ++r) {
      int m = mb + r;                       // 0..127 = b*2+g
      int b = m >> 1, g = m & 1;
      size_t xrow = (size_t)(b * TT + t) * 2 + g;
      const float* xp = xproj + xrow * G4;
      float pi = acc[i][0][r] + xp[0 * HH + hid] + bs0;
      float po = acc[i][1][r] + xp[1 * HH + hid] + bs1;
      float pf = acc[i][2][r] + xp[2 * HH + hid] + bs2;
      float pg = acc[i][3][r] + xp[3 * HH + hid] + bs3;
      float is = sigm(pi), os = sigm(po), fs = sigm(pf), gt = tanhf(pg);
      size_t ci = (size_t)m * HH + hid;
      float c_new = fs * c_state[ci] + is * gt;
      float h_new = os * tanhf(c_new);
      c_state[ci] = c_new;
      h_out[ci]   = f2bf(h_new);
      if (out_bf16) out_bf16[xrow * HH + hid] = f2bf(h_new);
      if (out_f32)  out_f32[(size_t)(b * TT + t) * (2 * HH) + g * HH + hid] = h_new;
    }
  }
}

// ------------------------------- host side ---------------------------------
extern "C" void kernel_launch(void* const* d_in, const int* in_sizes, int n_in,
                              void* d_out, int out_size, void* d_ws, size_t ws_size,
                              hipStream_t stream) {
  const float* feat  = (const float*)d_in[0];
  const float* sep_w = (const float*)d_in[1];
  const float* sep_b = (const float*)d_in[2];
  const float* Wx0 = (const float*)d_in[3];  const float* bx0 = (const float*)d_in[4];
  const float* Wh0 = (const float*)d_in[5];  const float* bh0 = (const float*)d_in[6];
  const float* Whh0= (const float*)d_in[7];  const float* bhh0= (const float*)d_in[8];
  const float* Wx1 = (const float*)d_in[9];  const float* bx1 = (const float*)d_in[10];
  const float* Wh1 = (const float*)d_in[11]; const float* bh1 = (const float*)d_in[12];
  const float* Whh1= (const float*)d_in[13]; const float* bhh1= (const float*)d_in[14];
  // d_in[15] = training (eval path assumed, == 0)

  // workspace carve-up (~369 MiB total)
  size_t off = 0;
  auto take = [&](size_t bytes) -> void* {
    void* p = (char*)d_ws + off;
    off += (bytes + 255) & ~(size_t)255;
    return p;
  };
  const size_t WELEMS = (size_t)G4 * HH;                 // 4096*1024
  unsigned short* wb[6];
  for (int i = 0; i < 6; ++i) wb[i] = (unsigned short*)take(WELEMS * 2);
  float* bsum0 = (float*)take(G4 * 4);
  float* bsum1 = (float*)take(G4 * 4);
  unsigned short* groups = (unsigned short*)take((size_t)ROWS_BT * HH * 2);
  unsigned short* hs0b   = (unsigned short*)take((size_t)ROWS_BT * HH * 2);
  float* xproj = (float*)take((size_t)ROWS_BT * G4 * 4);
  unsigned short* hst[2];
  hst[0] = (unsigned short*)take((size_t)ROWS_ST * HH * 2);
  hst[1] = (unsigned short*)take((size_t)ROWS_ST * HH * 2);
  float* cst = (float*)take((size_t)ROWS_ST * HH * 4);
  (void)ws_size; (void)in_sizes; (void)n_in; (void)out_size;

  // 1) weights -> bf16 (once per call; deterministic)
  const float* wsrc[6] = {Wx0, Wh0, Whh0, Wx1, Wh1, Whh1};
  for (int i = 0; i < 6; ++i)
    f32_to_bf16_kernel<<<2048, 256, 0, stream>>>(wsrc[i], wb[i], (int)WELEMS);
  add_bias_kernel<<<(G4 + 255) / 256, 256, 0, stream>>>(bh0, bhh0, bsum0, G4);
  add_bias_kernel<<<(G4 + 255) / 256, 256, 0, stream>>>(bh1, bhh1, bsum1, G4);

  // 2) separation + grouping
  sep_group_kernel<<<BB * TT, 256, 0, stream>>>(feat, sep_w, sep_b, groups);

  const int proj_blocks = (ROWS_BT / 64) * (G4 / 64) / 4;   // 4096
  const int step_blocks = (ROWS_ST / 32) * (HH / 16) / 4;   // 64
  const int state_words_h = ROWS_ST * HH / 2;               // bf16 state as u32
  const int state_words_c = ROWS_ST * HH;

  // ---- layer 0 ----
  gemm_proj_kernel<<<proj_blocks, 128, 0, stream>>>(groups, wb[0], bx0, xproj);
  zero_u32_kernel<<<256, 256, 0, stream>>>((unsigned int*)hst[0], state_words_h);
  zero_u32_kernel<<<256, 256, 0, stream>>>((unsigned int*)cst, state_words_c);
  for (int t = 0; t < TT; ++t)
    lstm_step_kernel<<<step_blocks, 128, 0, stream>>>(
        xproj, wb[1], wb[2], bsum0, hst[t & 1], hst[(t + 1) & 1], cst,
        hs0b, (float*)nullptr, t);

  // ---- layer 1 ----
  gemm_proj_kernel<<<proj_blocks, 128, 0, stream>>>(hs0b, wb[3], bx1, xproj);
  zero_u32_kernel<<<256, 256, 0, stream>>>((unsigned int*)hst[0], state_words_h);
  zero_u32_kernel<<<256, 256, 0, stream>>>((unsigned int*)cst, state_words_c);
  for (int t = 0; t < TT; ++t)
    lstm_step_kernel<<<step_blocks, 128, 0, stream>>>(
        xproj, wb[4], wb[5], bsum1, hst[t & 1], hst[(t + 1) & 1], cst,
        (unsigned short*)nullptr, (float*)d_out, t);
}
